// KernelizedLinearAttention_17978733101387
// MI455X (gfx1250) — matrix operand order repaired and checked by reading, not compiled, not run on hardware
//
#include <hip/hip_runtime.h>
#include <hip/hip_bf16.h>
#include <math.h>
#include <stdint.h>

// ---------------------------------------------------------------------------
// KernelizedLinearAttention for MI455X (gfx1250): three bf16 WMMA GEMMs with
// double-buffered LDS, GLOBAL_LOAD_ASYNC_TO_LDS staging and software-pipelined
// transpose staging.
//   phi  = SCALE*cos(x @ omega^T + bias)            [32768 x 256]  (+ phi^T)
//   S    = phi^T @ x   (per batch, stored S^T)      [8][1024 x 256]
//   out  = (phi @ S) / (phi @ Z + eps)              [8][4096 x 1024]
// ---------------------------------------------------------------------------

typedef __attribute__((ext_vector_type(16))) __bf16 v16bf;
typedef __attribute__((ext_vector_type(8)))  __bf16 v8bf;
typedef __attribute__((ext_vector_type(8)))  float  v8f;

#define BLK_THREADS 256          // 8 waves (wave32)
#define TILE_M      128
#define TILE_N      128
#define TILE_K      32
#define LDS_STRIDE  40           // 32 + 8 pad; 80B rows keep 16B alignment
#define TILE_ELEMS  (128 * LDS_STRIDE)

#define BB 8
#define LL 4096
#define DD 1024
#define MM 256
#define PHI_SCALE 0.08838834764831843f   // sqrt(2/256)

#define MODE_PHI 0
#define MODE_ST  1
#define MODE_OUT 2

// ---------------- CDNA5 async global->LDS copy (ASYNCcnt tracked) ----------
static __device__ __forceinline__ void async_copy_b128(uint32_t lds_byte,
                                                       const void* gsrc) {
  asm volatile("global_load_async_to_lds_b128 %0, %1, off"
               :: "v"(lds_byte), "v"((unsigned long long)(uintptr_t)gsrc)
               : "memory");
}
static __device__ __forceinline__ void wait_asynccnt0() {
  asm volatile("s_wait_asynccnt 0x0" ::: "memory");
}

// ---------------- fragment helpers (ISA 7.12.2 layouts) --------------------
static __device__ __forceinline__ v16bf frag_from_halves(v8bf lo, v8bf hi) {
  union { v16bf v; v8bf h[2]; } u;
  u.h[0] = lo; u.h[1] = hi;
  return u.v;
}

// A fragment (16x32): lanes 0-15 -> K 0..7/16..23, lanes 16-31 -> K 8..15/24..31
static __device__ __forceinline__ v16bf load_frag_a(const __bf16* At, int m0, int lane) {
  const int r = m0 + (lane & 15);
  const int h = lane >> 4;
  const __bf16* p = At + r * LDS_STRIDE + h * 8;
  v8bf lo = *(const v8bf*)p;
  v8bf hi = *(const v8bf*)(p + 16);
  return frag_from_halves(lo, hi);
}

// B fragment (32x16) from N-major tile Bt[n][k]: lanes 0-15 -> K 0..7/16..23, lanes 16-31 -> K 8..15/24..31
static __device__ __forceinline__ v16bf load_frag_b(const __bf16* Bt, int n0, int lane) {
  const int n = n0 + (lane & 15);
  const int h = lane >> 4;
  const __bf16* p = Bt + n * LDS_STRIDE + h * 8;
  v8bf lo = *(const v8bf*)p;
  v8bf hi = *(const v8bf*)(p + 16);
  return frag_from_halves(lo, hi);
}

// ---------------- staging: 128(mn) x 32(k) tiles ---------------------------
// mn-major global source: async DMA straight into LDS, no VGPR staging.
static __device__ __forceinline__ void stage_async(__bf16* dst, const __bf16* src,
                                                   int ld, int tid) {
  const int r = tid >> 1;          // 0..127
  const int h = tid & 1;           // 16-elem half of the 32-wide k slab
  const __bf16* g = src + (size_t)r * ld + h * 16;
  const uint32_t l = (uint32_t)(uintptr_t)(dst + r * LDS_STRIDE + h * 16);
  async_copy_b128(l, g);
  async_copy_b128(l + 16, g + 8);
}

// k-major global source: software pipeline (load regs early, LDS-store late).
struct TX { v8bf a, b; };
static __device__ __forceinline__ TX tx_load(const __bf16* src, int ld, int tid) {
  const int k = tid >> 3;          // 0..31
  const int q = tid & 7;           // 16-col group of the 128-wide tile
  const __bf16* p = src + (size_t)k * ld + q * 16;
  TX t;
  t.a = *(const v8bf*)p;
  t.b = *(const v8bf*)(p + 8);
  return t;
}
static __device__ __forceinline__ void tx_store(__bf16* dst, TX t, int tid) {
  const int k = tid >> 3;
  const int q = tid & 7;
  __bf16 tmp[16] __attribute__((aligned(16)));
  *(v8bf*)tmp       = t.a;
  *(v8bf*)(tmp + 8) = t.b;
  #pragma unroll
  for (int j = 0; j < 16; ++j)
    dst[(q * 16 + j) * LDS_STRIDE + k] = tmp[j];
}

struct GemmP {
  const __bf16* A; long long aBat; int lda;
  const __bf16* B; long long bBat; int ldb;
  int K;
  // MODE_PHI:
  const float* bias; float* Z; __bf16* phi; __bf16* phiT;
  // MODE_ST:
  __bf16* st;
  // MODE_OUT:
  const float* den; float* out;
};

template<int MODE, bool AK, bool BK>
__global__ void __launch_bounds__(BLK_THREADS)
gemm_wmma(GemmP p) {
  __shared__ __align__(16) __bf16 As[2][TILE_ELEMS];
  __shared__ __align__(16) __bf16 Bs[2][TILE_ELEMS];

  const int tid  = threadIdx.x;
  const int lane = tid & 31;
  const int w    = tid >> 5;        // 0..7
  const int wr   = w >> 1;          // row group: 32 rows each
  const int wc   = w & 1;           // col group: 64 cols each
  const int row0 = blockIdx.y * TILE_M;
  const int col0 = blockIdx.x * TILE_N;
  const int b    = blockIdx.z;

  const __bf16* Ab = p.A + (size_t)b * p.aBat;
  const __bf16* Bb = p.B + (size_t)b * p.bBat;

  v8f acc[2][4];
  #pragma unroll
  for (int i = 0; i < 2; ++i)
    #pragma unroll
    for (int j = 0; j < 4; ++j)
      #pragma unroll
      for (int v = 0; v < 8; ++v) acc[i][j][v] = 0.0f;

  const int ksteps = p.K / TILE_K;
  TX txa, txb;

  // ---- prologue: stage K-slab 0 into buffer 0 ----
  if (AK) txa = tx_load(Ab + row0, p.lda, tid);
  else    stage_async(As[0], Ab + (size_t)row0 * p.lda, p.lda, tid);
  if (BK) txb = tx_load(Bb + col0, p.ldb, tid);
  else    stage_async(Bs[0], Bb + (size_t)col0 * p.ldb, p.ldb, tid);
  if (AK) tx_store(As[0], txa, tid);
  if (BK) tx_store(Bs[0], txb, tid);
  wait_asynccnt0();
  __syncthreads();

  for (int kt = 0; kt < ksteps; ++kt) {
    const int cur = kt & 1;
    const int nxt = cur ^ 1;
    const bool hasNext = (kt + 1) < ksteps;

    // issue next slab's global traffic before computing this one
    if (hasNext) {
      const int k1 = (kt + 1) * TILE_K;
      if (AK) txa = tx_load(Ab + (size_t)k1 * p.lda + row0, p.lda, tid);
      else    stage_async(As[nxt], Ab + (size_t)row0 * p.lda + k1, p.lda, tid);
      if (BK) txb = tx_load(Bb + (size_t)k1 * p.ldb + col0, p.ldb, tid);
      else    stage_async(Bs[nxt], Bb + (size_t)col0 * p.ldb + k1, p.ldb, tid);
      if (kt + 2 < ksteps) {                       // global_prefetch_b8 for kt+2
        const int k2 = (kt + 2) * TILE_K;
        const __bf16* pa = AK ? Ab + (size_t)(k2 + (tid >> 3)) * p.lda + row0
                              : Ab + (size_t)(row0 + (tid >> 1)) * p.lda + k2;
        const __bf16* pb = BK ? Bb + (size_t)(k2 + (tid >> 3)) * p.ldb + col0
                              : Bb + (size_t)(col0 + (tid >> 1)) * p.ldb + k2;
        __builtin_prefetch(pa, 0, 1);
        __builtin_prefetch(pb, 0, 1);
      }
    }

    // ---- compute current slab: 8 WMMAs per wave per K-step ----
    const __bf16* Ac = As[cur];
    const __bf16* Bc = Bs[cur];
    v16bf a0 = load_frag_a(Ac, wr * 32,      lane);
    v16bf a1 = load_frag_a(Ac, wr * 32 + 16, lane);
    #pragma unroll
    for (int nt = 0; nt < 4; ++nt) {
      v16bf bf = load_frag_b(Bc, wc * 64 + nt * 16, lane);
      acc[0][nt] = __builtin_amdgcn_wmma_f32_16x16x32_bf16(
          false, a0, false, bf, (short)0, acc[0][nt], false, false);
      acc[1][nt] = __builtin_amdgcn_wmma_f32_16x16x32_bf16(
          false, a1, false, bf, (short)0, acc[1][nt], false, false);
    }

    // late LDS stores of the pipelined transpose regs (other buffer)
    if (hasNext) {
      if (AK) tx_store(As[nxt], txa, tid);
      if (BK) tx_store(Bs[nxt], txb, tid);
    }
    wait_asynccnt0();
    __syncthreads();
  }

  // ---- epilogues (C/D layout: lane = n + 16*(m>>3); VGPR v -> M = v+8*(lane>>4))
  const int nl      = lane & 15;
  const int h       = lane >> 4;
  const int colBase = col0 + wc * 64;

  if (MODE == MODE_PHI) {
    const int bidx = row0 / LL;               // 128-row blocks never straddle a batch
    #pragma unroll
    for (int rt = 0; rt < 2; ++rt) {
      const int mbase = row0 + wr * 32 + rt * 16 + 8 * h;
      const int lbase = mbase - bidx * LL;
      #pragma unroll
      for (int nt = 0; nt < 4; ++nt) {
        const int c = colBase + nt * 16 + nl;
        const float bias_c = p.bias[c];
        float partial = 0.0f;
        v8bf tp;
        #pragma unroll
        for (int v = 0; v < 8; ++v) {
          const float val = PHI_SCALE * __builtin_cosf(acc[rt][nt][v] + bias_c);
          p.phi[(size_t)(mbase + v) * MM + c] = (__bf16)val;
          tp[v] = (__bf16)val;
          partial += val;
        }
        // transposed copy phiT[b][m][l]: contiguous in v -> one 16B store
        *(v8bf*)(p.phiT + (size_t)bidx * MM * LL + (size_t)c * LL + lbase) = tp;
        atomicAdd(&p.Z[bidx * MM + c], partial);
      }
    }
  } else if (MODE == MODE_ST) {
    // output S[m][n=d] stored transposed: st[b][n*MM + m] (contiguous over v)
    #pragma unroll
    for (int rt = 0; rt < 2; ++rt) {
      const int mbase = row0 + wr * 32 + rt * 16 + 8 * h;
      #pragma unroll
      for (int nt = 0; nt < 4; ++nt) {
        const int n = colBase + nt * 16 + nl;
        v8bf o;
        #pragma unroll
        for (int v = 0; v < 8; ++v) o[v] = (__bf16)acc[rt][nt][v];
        *(v8bf*)(p.st + (size_t)b * DD * MM + (size_t)n * MM + mbase) = o;
      }
    }
  } else {  // MODE_OUT
    #pragma unroll
    for (int rt = 0; rt < 2; ++rt) {
      const int mbase = row0 + wr * 32 + rt * 16 + 8 * h;
      float inv[8];
      #pragma unroll
      for (int v = 0; v < 8; ++v)
        inv[v] = 1.0f / p.den[(size_t)b * LL + mbase + v];
      #pragma unroll
      for (int nt = 0; nt < 4; ++nt) {
        const int n = colBase + nt * 16 + nl;
        #pragma unroll
        for (int v = 0; v < 8; ++v)
          p.out[(size_t)b * LL * DD + (size_t)(mbase + v) * DD + n] = acc[rt][nt][v] * inv[v];
      }
    }
  }
}

// fp32 -> bf16 conversion of x and omega; zero Z accumulators
__global__ void convert_zero_kernel(const float* __restrict__ x,
                                    const float* __restrict__ om,
                                    __bf16* __restrict__ xb,
                                    __bf16* __restrict__ ob,
                                    float* __restrict__ Z,
                                    long long nx, long long no) {
  const long long i0 = (long long)blockIdx.x * blockDim.x + threadIdx.x;
  const long long stride = (long long)gridDim.x * blockDim.x;
  for (long long i = i0; i < nx; i += stride) xb[i] = (__bf16)x[i];
  for (long long i = i0; i < no; i += stride) ob[i] = (__bf16)om[i];
  if (i0 < BB * MM) Z[i0] = 0.0f;
}

// den[b][l] = eps + sum_m phi[b][l][m] * Z[b][m]
__global__ void den_kernel(const __bf16* __restrict__ phi,
                           const float* __restrict__ Z,
                           float* __restrict__ den) {
  const int r = blockIdx.x * blockDim.x + threadIdx.x;   // 0 .. B*L-1
  const int b = r >> 12;                                 // r / 4096
  const __bf16* pr = phi + (size_t)r * MM;
  const float*  z  = Z + b * MM;
  float s = 0.0f;
  #pragma unroll 4
  for (int m = 0; m < MM; m += 8) {
    v8bf t = *(const v8bf*)(pr + m);
    #pragma unroll
    for (int j = 0; j < 8; ++j) s += (float)t[j] * z[m + j];
  }
  den[r] = s + 1e-8f;
}

extern "C" void kernel_launch(void* const* d_in, const int* in_sizes, int n_in,
                              void* d_out, int out_size, void* d_ws, size_t ws_size,
                              hipStream_t stream) {
  (void)in_sizes; (void)n_in; (void)out_size; (void)ws_size;

  const float* x     = (const float*)d_in[0];   // [8,4096,1024]
  const float* omega = (const float*)d_in[1];   // [256,1024]
  const float* bias  = (const float*)d_in[2];   // [256]
  float* out = (float*)d_out;                   // [8,4096,1024]

  char* ws = (char*)d_ws;
  const size_t nx   = (size_t)BB * LL * DD;     // 33,554,432
  const size_t nom  = (size_t)MM * DD;          // 262,144
  const size_t nphi = (size_t)BB * LL * MM;     // 8,388,608
  const size_t nst  = (size_t)BB * DD * MM;     // 2,097,152

  size_t off = 0;
  __bf16* xb   = (__bf16*)(ws + off); off += nx   * 2;   // 64 MiB
  __bf16* ob   = (__bf16*)(ws + off); off += nom  * 2;   // 512 KiB
  __bf16* phi  = (__bf16*)(ws + off); off += nphi * 2;   // 16 MiB
  __bf16* phiT = (__bf16*)(ws + off); off += nphi * 2;   // 16 MiB
  __bf16* st   = (__bf16*)(ws + off); off += nst  * 2;   // 4 MiB
  float*  Z    = (float*) (ws + off); off += (size_t)BB * MM * 4;
  float*  den  = (float*) (ws + off); off += (size_t)BB * LL * 4;

  // 1) convert inputs to bf16, zero Z
  convert_zero_kernel<<<2048, 256, 0, stream>>>(x, omega, xb, ob, Z,
                                                (long long)nx, (long long)nom);

  // 2) phi = SCALE*cos(x @ omega^T + bias); also phi^T and Z column sums
  {
    GemmP p = {};
    p.A = xb; p.aBat = 0; p.lda = DD;          // [32768 x 1024] mn-major -> async
    p.B = ob; p.bBat = 0; p.ldb = DD;          // omega == Bt natural      -> async
    p.K = DD;
    p.bias = bias; p.Z = Z; p.phi = phi; p.phiT = phiT;
    dim3 grid(MM / TILE_N, (BB * LL) / TILE_M, 1);       // (2, 256, 1)
    gemm_wmma<MODE_PHI, false, false><<<grid, BLK_THREADS, 0, stream>>>(p);
  }

  // 3) den = phi @ Z + eps
  den_kernel<<<(BB * LL) / 256, 256, 0, stream>>>(phi, Z, den);

  // 4) S = phiT @ x per batch (K = L = 4096), stored transposed bf16 St[D][M]
  {
    GemmP p = {};
    p.A = phiT; p.aBat = (long long)MM * LL; p.lda = LL; // natural -> async
    p.B = xb;   p.bBat = (long long)LL * DD; p.ldb = DD; // k-major -> pipelined LDS transpose
    p.K = LL;
    p.st = st;
    dim3 grid(DD / TILE_N, MM / TILE_M, BB);             // (8, 2, 8)
    gemm_wmma<MODE_ST, false, true><<<grid, BLK_THREADS, 0, stream>>>(p);
  }

  // 5) out = (phi @ S) / den
  {
    GemmP p = {};
    p.A = phi; p.aBat = (long long)LL * MM; p.lda = MM;  // natural -> async
    p.B = st;  p.bBat = (long long)DD * MM; p.ldb = MM;  // St natural -> async
    p.K = MM;
    p.den = den; p.out = out;
    dim3 grid(DD / TILE_N, LL / TILE_M, BB);             // (8, 32, 8)
    gemm_wmma<MODE_OUT, false, false><<<grid, BLK_THREADS, 0, stream>>>(p);
  }
}
